// NeuralCDE_27281632264785
// MI455X (gfx1250) — matrix operand order repaired and checked by reading, not compiled, not run on hardware
//
#include <hip/hip_runtime.h>

#define BATCH   256
#define SEQL    100
#define DIN     32
#define HDIM    64
#define WDIM    128
#define OUTC    20
#define NSTEPS  100
#define BT      16      // batch tile per workgroup (== WMMA M)

typedef __attribute__((ext_vector_type(16))) __bf16 v16bf;
typedef __attribute__((ext_vector_type(8)))  float  v8f;

union FragB { v16bf v; uint4 q[2]; };

#if __has_builtin(__builtin_amdgcn_tanhf)
__device__ __forceinline__ float fast_tanh(float x) { return __builtin_amdgcn_tanhf(x); }
#else
__device__ __forceinline__ float fast_tanh(float x) { return tanhf(x); }
#endif

// RK6 tableau
__constant__ float ACO[6][5] = {
  {0.f, 0.f, 0.f, 0.f, 0.f},
  {0.161f, 0.f, 0.f, 0.f, 0.f},
  {-0.008480655492356989f, 0.335480655492357f, 0.f, 0.f, 0.f},
  {2.8971530571054935f, -6.359448489975075f, 4.3622954328695815f, 0.f, 0.f},
  {5.325864828439257f, -11.748883564062828f, 7.4955393428898365f, -0.09249506636175525f, 0.f},
  {5.86145544294642f, -12.92096931784711f, 8.159367898576159f, -0.071584973281401f, -0.028269050394068383f}
};
__constant__ float CCO[6] = {0.f, 0.161f, 0.327f, 0.9f, 0.9800255409045097f, 1.f};
__constant__ float BWO[6] = {0.09646076681806523f, 0.01f, 0.4798896504144996f,
                             1.379008574103742f, -3.290069515436081f, 2.324710524099774f};

// ---------------------------------------------------------------------------
// Pack W (N_out x K_in, row-major f32) transposed into bf16 WMMA B-fragments.
// frag[(kt*NT + nt)*512 + lane*16 + j] = bf16(W[n][k]),
//   n = nt*16 + (lane&15),  k = kt*32 + 16*(lane>>4) + j
// so a wave loads its B tile with two coalesced 16B loads per lane.
// ---------------------------------------------------------------------------
__global__ __launch_bounds__(256) void pack_bfrag(const float* __restrict__ Wm,
                                                  int K, int N,
                                                  __bf16* __restrict__ frag) {
  const int NT = N >> 4;
  const int total = (K >> 5) * NT * 512;
  for (int idx = blockIdx.x * blockDim.x + threadIdx.x; idx < total;
       idx += gridDim.x * blockDim.x) {
    int j    = idx & 15;
    int lane = (idx >> 4) & 31;
    int tile = idx >> 9;
    int nt   = tile % NT;
    int kt   = tile / NT;
    int n = nt * 16 + (lane & 15);
    int k = kt * 32 + 16 * (lane >> 4) + j;
    frag[idx] = (__bf16)Wm[(size_t)n * K + k];
  }
}

// ---------------------------------------------------------------------------
// y0 = iW2 @ relu(iW1 @ relu(iW0 @ x[0] + ib0) + ib1) + ib2   (tiny, scalar)
// ---------------------------------------------------------------------------
__global__ __launch_bounds__(128) void init_y0(const float* __restrict__ xs,
    const float* __restrict__ iW0, const float* __restrict__ ib0,
    const float* __restrict__ iW1, const float* __restrict__ ib1,
    const float* __restrict__ iW2, const float* __restrict__ ib2,
    float* __restrict__ y0g) {
  __shared__ float x0[DIN];
  __shared__ float h0[WDIM];
  __shared__ float h1[WDIM];
  const int b = blockIdx.x, t = threadIdx.x;
  if (t < DIN) x0[t] = xs[(size_t)b * SEQL * DIN + t];
  __syncthreads();
  float a = ib0[t];
  for (int j = 0; j < DIN; ++j) a += iW0[t * DIN + j] * x0[j];
  h0[t] = fmaxf(a, 0.f);
  __syncthreads();
  a = ib1[t];
  for (int j = 0; j < WDIM; ++j) a += iW1[t * WDIM + j] * h0[j];
  h1[t] = fmaxf(a, 0.f);
  __syncthreads();
  if (t < HDIM) {
    a = ib2[t];
    for (int j = 0; j < WDIM; ++j) a += iW2[t * WDIM + j] * h1[j];
    y0g[(size_t)b * HDIM + t] = a;
  }
}

// ---------------------------------------------------------------------------
// Persistent RK6 integrator: one WG = 16 trajectories, 8 waves, full 600-stage
// chain in LDS.  All matmuls via v_wmma_f32_16x16x32_bf16.
// ---------------------------------------------------------------------------
__global__ void __launch_bounds__(256)
cde_main(const float* __restrict__ xs,
         const float* __restrict__ fb0, const float* __restrict__ fb1,
         const float* __restrict__ fb2,
         const float* __restrict__ lW,  const float* __restrict__ lb,
         const __bf16* __restrict__ w0f,
         const __bf16* __restrict__ w1f,
         const __bf16* __restrict__ w2f,
         const float* __restrict__ y0g,
         float* __restrict__ out) {
  __shared__ __align__(16) float   Y[BT * HDIM];        // ODE state
  __shared__ __align__(16) float   kbuf[6][BT * HDIM];  // RK stage slopes
  __shared__ __align__(16) __bf16  Abuf[BT * HDIM];     // stage input, bf16
  __shared__ __align__(16) __bf16  H1[BT * WDIM];       // relu(fW0 y)
  __shared__ __align__(16) __bf16  H2[BT * WDIM];       // relu(fW1 h1)
  __shared__ __align__(16) float   dX[BT * DIN];        // spline derivative

  const int tid  = threadIdx.x;
  const int lane = tid & 31;
  const int wv   = tid >> 5;        // 8 waves
  const int half = lane >> 4;       // 0: rows r, 1: rows r+8 (C layout)
  const int ln   = lane & 15;       // N within tile / M row for A
  const int bg   = blockIdx.x * BT;

  // biases are step/stage invariant: pin them in registers for the whole run
  const float bias0 = fb0[wv * 16 + ln];
  const float bias1 = fb1[wv * 16 + ln];
  float bias2[16];
  #pragma unroll
  for (int q = 0; q < 16; ++q) bias2[q] = fb2[(wv * 16 + q) * 16 + ln];

  for (int i = tid; i < BT * HDIM; i += 256) Y[i] = y0g[(size_t)bg * HDIM + i];
  __syncthreads();

  const float hstep = 1.0f / (float)NSTEPS;
  const float gs = (float)(SEQL - 1);            // 1/dt of the linspace grid

  for (int step = 0; step < NSTEPS; ++step) {
    const float tbase = (float)step * hstep;

    for (int s = 0; s < 6; ++s) {
      // ---- stage input ytmp = y + h*sum(a_sp * k_p) -> bf16; and dX(t_s) ----
      for (int i = tid; i < BT * HDIM; i += 256) {
        float acc = Y[i];
        for (int p = 0; p < s; ++p) acc += hstep * ACO[s][p] * kbuf[p][i];
        Abuf[i] = (__bf16)acc;
      }
      const float tstage = tbase + CCO[s] * hstep;
      for (int i = tid; i < BT * DIN; i += 256) {
        const int m = i >> 5, d = i & 31;
        const float* xb = xs + (size_t)(bg + m) * (SEQL * DIN);
        int ii = (int)(tstage * gs);
        ii = ii < 0 ? 0 : (ii > SEQL - 2 ? SEQL - 2 : ii);
        const float ss = tstage - (float)ii / gs;
        const float xi = xb[ii * DIN + d];
        const float xp = xb[(ii + 1) * DIN + d];
        const float bdi = (xp - xi) * gs;                       // slope = f1
        float f0 = bdi;
        if (ii > 0) f0 = (xi - xb[(ii - 1) * DIN + d]) * gs;
        const float ccv = 2.0f * (bdi - f0) * gs;               // (3s-2f0-f1)/dt
        const float cdv = (f0 - bdi) * gs * gs;                 // (f0+f1-2s)/dt^2
        dX[i] = f0 + 2.0f * ccv * ss + 3.0f * cdv * ss * ss;
      }
      __syncthreads();

      // ---- GEMM1: H1 = relu(ytmp(16x64) x W0^T(64x128) + fb0) ----
      {
        const int nt = wv;
        const int n = nt * 16 + ln;
        FragB af[2], bf[2];
        #pragma unroll
        for (int kt = 0; kt < 2; ++kt) {
          const uint4* ap = (const uint4*)&Abuf[ln * HDIM + kt * 32 + half * 16];
          af[kt].q[0] = ap[0]; af[kt].q[1] = ap[1];   // K run [c,c+16)
          const uint4* bp = (const uint4*)&w0f[(size_t)((kt * 8 + nt) * 32 + lane) * 16];
          bf[kt].q[0] = bp[0]; bf[kt].q[1] = bp[1];
        }
        v8f c;
        #pragma unroll
        for (int r = 0; r < 8; ++r) c[r] = bias0;
        #pragma unroll
        for (int kt = 0; kt < 2; ++kt)
          c = __builtin_amdgcn_wmma_f32_16x16x32_bf16(false, af[kt].v, false, bf[kt].v,
                                                      (short)0, c, false, false);
        #pragma unroll
        for (int r = 0; r < 8; ++r)
          H1[(r + 8 * half) * WDIM + n] = (__bf16)fmaxf(c[r], 0.f);
      }
      __syncthreads();

      // ---- GEMM2: H2 = relu(H1(16x128) x W1^T(128x128) + fb1) ----
      {
        const int nt = wv;
        const int n = nt * 16 + ln;
        FragB af[4], bf[4];
        #pragma unroll
        for (int kt = 0; kt < 4; ++kt) {
          const uint4* ap = (const uint4*)&H1[ln * WDIM + kt * 32 + half * 16];
          af[kt].q[0] = ap[0]; af[kt].q[1] = ap[1];
          const uint4* bp = (const uint4*)&w1f[(size_t)((kt * 8 + nt) * 32 + lane) * 16];
          bf[kt].q[0] = bp[0]; bf[kt].q[1] = bp[1];
        }
        v8f c;
        #pragma unroll
        for (int r = 0; r < 8; ++r) c[r] = bias1;
        #pragma unroll
        for (int kt = 0; kt < 4; ++kt)
          c = __builtin_amdgcn_wmma_f32_16x16x32_bf16(false, af[kt].v, false, bf[kt].v,
                                                      (short)0, c, false, false);
        #pragma unroll
        for (int r = 0; r < 8; ++r)
          H2[(r + 8 * half) * WDIM + n] = (__bf16)fmaxf(c[r], 0.f);
      }
      __syncthreads();

      // ---- GEMM3 + contraction: k_s[m][hh] = sum_dd tanh(..)[m][hh*32+dd]*dX[m][dd]
      // wave wv owns hh = wv*8 .. wv*8+7 (two 16-wide N tiles each); the 16x2048
      // tanh output is consumed in-register, never materialized.  The A
      // fragments are shared by all 16 N-tiles: load them ONCE per stage.
      {
        FragB af[4];
        #pragma unroll
        for (int kt = 0; kt < 4; ++kt) {
          const uint4* ap = (const uint4*)&H2[ln * WDIM + kt * 32 + half * 16];
          af[kt].q[0] = ap[0]; af[kt].q[1] = ap[1];
        }
        float kacc[8];
        for (int hq = 0; hq < 8; ++hq) {
          const int hh = wv * 8 + hq;
          #pragma unroll
          for (int r = 0; r < 8; ++r) kacc[r] = 0.f;
          if (hq + 1 < 8)   // stream next weight tile into cache
            __builtin_prefetch(&w2f[(size_t)(((hh + 1) * 2) * 32 + lane) * 16], 0, 1);
          #pragma unroll
          for (int nh = 0; nh < 2; ++nh) {
            const int nt = hh * 2 + nh;
            const int dd = ln + 16 * nh;
            FragB bf[4];
            #pragma unroll
            for (int kt = 0; kt < 4; ++kt) {
              const uint4* bp = (const uint4*)&w2f[(size_t)((kt * 128 + nt) * 32 + lane) * 16];
              bf[kt].q[0] = bp[0]; bf[kt].q[1] = bp[1];
            }
            v8f c;
            #pragma unroll
            for (int r = 0; r < 8; ++r) c[r] = bias2[hq * 2 + nh];
            #pragma unroll
            for (int kt = 0; kt < 4; ++kt)
              c = __builtin_amdgcn_wmma_f32_16x16x32_bf16(false, af[kt].v, false, bf[kt].v,
                                                          (short)0, c, false, false);
            #pragma unroll
            for (int r = 0; r < 8; ++r) {
              const int m = r + 8 * half;
              float v = fast_tanh(c[r]) * dX[m * DIN + dd];
              v += __shfl_xor(v, 1, 32);      // butterfly over the 16 N-lanes
              v += __shfl_xor(v, 2, 32);
              v += __shfl_xor(v, 4, 32);
              v += __shfl_xor(v, 8, 32);
              kacc[r] += v;
            }
          }
          if (ln == 0) {                      // lanes 0 (m=r) and 16 (m=r+8)
            #pragma unroll
            for (int r = 0; r < 8; ++r)
              kbuf[s][(r + 8 * half) * HDIM + hh] = kacc[r];
          }
        }
      }
      __syncthreads();
    } // stages

    // ---- y += h * sum(b_i * k_i) ----
    for (int i = tid; i < BT * HDIM; i += 256) {
      float acc = Y[i];
      #pragma unroll
      for (int p = 0; p < 6; ++p) acc += hstep * BWO[p] * kbuf[p][i];
      Y[i] = acc;
    }
    __syncthreads();
  } // steps

  // ---- out = softmax(lW @ yT + lb) ----
  if (tid < BT) {
    const int m = tid;
    float z[OUTC];
    float zmax = -3.4e38f;
    for (int o = 0; o < OUTC; ++o) {
      float acc = lb[o];
      for (int j = 0; j < HDIM; ++j) acc += lW[o * HDIM + j] * Y[m * HDIM + j];
      z[o] = acc;
      zmax = fmaxf(zmax, acc);
    }
    float se = 0.f;
    for (int o = 0; o < OUTC; ++o) { z[o] = __expf(z[o] - zmax); se += z[o]; }
    const float inv = 1.f / se;
    for (int o = 0; o < OUTC; ++o) out[(size_t)(bg + m) * OUTC + o] = z[o] * inv;
  }
}

// ---------------------------------------------------------------------------
extern "C" void kernel_launch(void* const* d_in, const int* in_sizes, int n_in,
                              void* d_out, int out_size, void* d_ws, size_t ws_size,
                              hipStream_t stream) {
  (void)in_sizes; (void)n_in; (void)out_size; (void)ws_size;
  const float* xs  = (const float*)d_in[0];
  const float* iW0 = (const float*)d_in[1];
  const float* ib0 = (const float*)d_in[2];
  const float* iW1 = (const float*)d_in[3];
  const float* ib1 = (const float*)d_in[4];
  const float* iW2 = (const float*)d_in[5];
  const float* ib2 = (const float*)d_in[6];
  const float* fW0 = (const float*)d_in[7];
  const float* fb0 = (const float*)d_in[8];
  const float* fW1 = (const float*)d_in[9];
  const float* fb1 = (const float*)d_in[10];
  const float* fW2 = (const float*)d_in[11];
  const float* fb2 = (const float*)d_in[12];
  const float* lW  = (const float*)d_in[13];
  const float* lb  = (const float*)d_in[14];

  // workspace layout (bytes): w0f 16K | w1f 32K | w2f 512K | y0 64K  (~624KB)
  __bf16* w0f = (__bf16*)d_ws;                 // 16 tiles * 512
  __bf16* w1f = w0f + 16 * 512;                // 32 tiles * 512
  __bf16* w2f = w1f + 32 * 512;                // 512 tiles * 512
  float*  y0g = (float*)(w2f + 512 * 512);     // 256 * 64 f32

  pack_bfrag<<<16, 256, 0, stream>>>(fW0, HDIM, WDIM, w0f);
  pack_bfrag<<<32, 256, 0, stream>>>(fW1, WDIM, WDIM, w1f);
  pack_bfrag<<<256, 256, 0, stream>>>(fW2, WDIM, HDIM * DIN, w2f);
  init_y0<<<BATCH, 128, 0, stream>>>(xs, iW0, ib0, iW1, ib1, iW2, ib2, y0g);

  cde_main<<<BATCH / BT, 256, 0, stream>>>(xs, fb0, fb1, fb2, lW, lb,
                                           w0f, w1f, w2f, y0g, (float*)d_out);
}
